// Attention_7670811591272
// MI455X (gfx1250) — matrix operand order repaired and checked by reading, not compiled, not run on hardware
//
#include <hip/hip_runtime.h>

typedef _Float16 half_t;
typedef __attribute__((ext_vector_type(16))) _Float16 v16h;
typedef __attribute__((ext_vector_type(8)))  _Float16 v8h;
typedef __attribute__((ext_vector_type(8)))  float    v8f;
typedef __attribute__((ext_vector_type(4)))  float    v4f;

#define WMMA_F32_F16(a, b, c) \
  __builtin_amdgcn_wmma_f32_16x16x32_f16(false, (a), false, (b), (short)0, (c), false, false)

// Use builtin presence only as a capability probe for the gfx1250 async-to-LDS path;
// the actual instruction is emitted via inline asm (ISA 15.18.3, opcode 98).
#if defined(__has_builtin)
#if __has_builtin(__builtin_amdgcn_global_load_async_to_lds_b128)
#define HAVE_ASYNC_TO_LDS 1
#endif
#endif

static __device__ __forceinline__ int lane_id() { return (int)(threadIdx.x & 31u); }

// ---------------- Fragment loaders (contiguous k order, same for A and B) ----------------
// A fragment 16x32 f16: lane L holds row m0+(L&15); lanes 0-15: K kk..kk+15,
// lanes 16-31: K kk+16..kk+31; 2 halves per VGPR, low bits = lower K.
static __device__ __forceinline__ v16h load_a_frag(const half_t* __restrict__ src, int ld,
                                                   int m0, int kk) {
  const int lane = lane_id();
  const half_t* p = src + (size_t)(m0 + (lane & 15)) * ld + kk + ((lane & 16) ? 16 : 0);
  v16h a;
#pragma unroll
  for (int i = 0; i < 8; ++i) { a[i] = p[i]; a[8 + i] = p[8 + i]; }
  return a;
}
static __device__ __forceinline__ v16h load_a_frag(const float* __restrict__ src, int ld,
                                                   int m0, int kk) {
  const int lane = lane_id();
  const float* p = src + (size_t)(m0 + (lane & 15)) * ld + kk + ((lane & 16) ? 16 : 0);
  v16h a;
#pragma unroll
  for (int i = 0; i < 8; ++i) { a[i] = (half_t)p[i]; a[8 + i] = (half_t)p[8 + i]; }
  return a;
}

// B fragment 32x16: lane L holds column n0+(L&15); lanes 0-15: K kk..kk+15,
// lanes 16-31: K kk+16..kk+31. Source row-major [n][k] -> 16 contiguous halves.
static __device__ __forceinline__ v16h load_b_rowk(const half_t* __restrict__ src, int ld,
                                                   int n0, int kk) {
  const int lane = lane_id();
  const half_t* p = src + (size_t)(n0 + (lane & 15)) * ld + kk + ((lane & 16) ? 16 : 0);
  v16h b;
#pragma unroll
  for (int i = 0; i < 16; ++i) b[i] = p[i];
  return b;
}

// B fragment from a padded LDS tile Vt[n][k] (row stride ldh halves): contiguous in k.
static __device__ __forceinline__ v16h load_b_lds(const half_t* Vt, int ldh, int n0) {
  const int lane = lane_id();
  const half_t* p = Vt + (size_t)(n0 + (lane & 15)) * ldh + ((lane & 16) ? 16 : 0);
  v16h b;
#pragma unroll
  for (int i = 0; i < 16; ++i) b[i] = p[i];
  return b;
}

// ---------------- Kernel 0: fp32 -> f16 convert (8 elems/thread) ----------------
__global__ __launch_bounds__(256) void cvt_f32_f16_kernel(const float* __restrict__ in,
                                                          half_t* __restrict__ out, int n8) {
  const int i = (int)(blockIdx.x * blockDim.x + threadIdx.x);
  if (i >= n8) return;
  const v4f* p = (const v4f*)in + (size_t)i * 2;
  const v4f x = p[0], y = p[1];
  v8h o;
#pragma unroll
  for (int j = 0; j < 4; ++j) { o[j] = (half_t)x[j]; o[4 + j] = (half_t)y[j]; }
  *(v8h*)(out + (size_t)i * 8) = o;
}

// ---------------- Kernel 1/4: projection GEMM  Y = X @ W^T + bias (all f16 in) ----------
// X: [4096,1024] f16, W: [1024,1024] f16 row-major (out,in), Y: [4096,1024] TY.
// 8 waves (4 row x 2 col), wave tile 32x32 -> block tile 128x64, K-step 32.
template <typename TY>
__global__ __launch_bounds__(256) void proj_gemm_kernel(const half_t* __restrict__ X,
                                                        const half_t* __restrict__ W,
                                                        const float* __restrict__ bias,
                                                        TY* __restrict__ Y) {
  const int wid = (int)(threadIdx.x >> 5);
  const int m0 = (int)blockIdx.y * 128 + (wid & 3) * 32;
  const int n0 = (int)blockIdx.x * 64 + (wid >> 2) * 32;
  v8f c00 = {}, c01 = {}, c10 = {}, c11 = {};
  for (int kk = 0; kk < 1024; kk += 32) {
    v16h a0 = load_a_frag(X, 1024, m0, kk);
    v16h a1 = load_a_frag(X, 1024, m0 + 16, kk);
    v16h b0 = load_b_rowk(W, 1024, n0, kk);
    v16h b1 = load_b_rowk(W, 1024, n0 + 16, kk);
    c00 = WMMA_F32_F16(a0, b0, c00);
    c01 = WMMA_F32_F16(a0, b1, c01);
    c10 = WMMA_F32_F16(a1, b0, c10);
    c11 = WMMA_F32_F16(a1, b1, c11);
  }
  const int lane = lane_id();
  const int rb = (lane & 16) ? 8 : 0;
  const int col = n0 + (lane & 15);
  const float bv0 = bias[col];
  const float bv1 = bias[col + 16];
#pragma unroll
  for (int i = 0; i < 8; ++i) {
    const size_t r0 = (size_t)(m0 + rb + i) * 1024;
    const size_t r1 = (size_t)(m0 + 16 + rb + i) * 1024;
    Y[r0 + col]      = (TY)(c00[i] + bv0);
    Y[r0 + col + 16] = (TY)(c01[i] + bv1);
    Y[r1 + col]      = (TY)(c10[i] + bv0);
    Y[r1 + col + 16] = (TY)(c11[i] + bv1);
  }
}

// ---------------- Kernel 2: fused scores + softmax ----------------
__global__ __launch_bounds__(256) void attn_softmax_kernel(const half_t* __restrict__ q,
                                                           const half_t* __restrict__ k,
                                                           float* __restrict__ scores) {
  const int bh = (int)blockIdx.y;               // b*16 + h
  const int b = bh >> 4, h = bh & 15;
  const int s0 = (int)blockIdx.x * 16;
  const int w = (int)(threadIdx.x >> 5);
  const int lane = lane_id();
  const half_t* qh = q + (size_t)b * 2048 * 1024 + h * 64;
  const half_t* kh = k + (size_t)b * 2048 * 1024 + h * 64;

  const v16h a0 = load_a_frag(qh, 1024, s0, 0);
  const v16h a1 = load_a_frag(qh, 1024, s0, 32);

  v8f acc[16];
#pragma unroll
  for (int tt = 0; tt < 16; ++tt) {
    const int t0 = w * 256 + tt * 16;
    v16h b0 = load_b_rowk(kh, 1024, t0, 0);
    v16h b1 = load_b_rowk(kh, 1024, t0, 32);
    v8f c = {};
    c = WMMA_F32_F16(a0, b0, c);
    c = WMMA_F32_F16(a1, b1, c);
#pragma unroll
    for (int i = 0; i < 8; ++i) c[i] *= 0.125f;   // 1/sqrt(dk)
    acc[tt] = c;
  }

  __shared__ float wred[8][16];
  const int rb = (lane & 16) ? 8 : 0;

  // ---- row max ----
  float pm[8];
#pragma unroll
  for (int i = 0; i < 8; ++i) {
    float m = acc[0][i];
#pragma unroll
    for (int tt = 1; tt < 16; ++tt) m = fmaxf(m, acc[tt][i]);
    pm[i] = m;
  }
#pragma unroll
  for (int off = 1; off < 16; off <<= 1)
#pragma unroll
    for (int i = 0; i < 8; ++i) pm[i] = fmaxf(pm[i], __shfl_xor(pm[i], off, 32));
  if (lane == 0)
    for (int i = 0; i < 8; ++i) wred[w][i] = pm[i];
  if (lane == 16)
    for (int i = 0; i < 8; ++i) wred[w][8 + i] = pm[i];
  __syncthreads();
  float rmax[8];
#pragma unroll
  for (int i = 0; i < 8; ++i) {
    float m = wred[0][rb + i];
#pragma unroll
    for (int ww = 1; ww < 8; ++ww) m = fmaxf(m, wred[ww][rb + i]);
    rmax[i] = m;
  }
  __syncthreads();

  // ---- exp + row sum ----
  float ps[8] = {0.f, 0.f, 0.f, 0.f, 0.f, 0.f, 0.f, 0.f};
#pragma unroll
  for (int tt = 0; tt < 16; ++tt)
#pragma unroll
    for (int i = 0; i < 8; ++i) {
      const float e = __expf(acc[tt][i] - rmax[i]);
      acc[tt][i] = e;
      ps[i] += e;
    }
#pragma unroll
  for (int off = 1; off < 16; off <<= 1)
#pragma unroll
    for (int i = 0; i < 8; ++i) ps[i] += __shfl_xor(ps[i], off, 32);
  if (lane == 0)
    for (int i = 0; i < 8; ++i) wred[w][i] = ps[i];
  if (lane == 16)
    for (int i = 0; i < 8; ++i) wred[w][8 + i] = ps[i];
  __syncthreads();
  float rinv[8];
#pragma unroll
  for (int i = 0; i < 8; ++i) {
    float s = wred[0][rb + i];
#pragma unroll
    for (int ww = 1; ww < 8; ++ww) s += wred[ww][rb + i];
    rinv[i] = 1.0f / s;
  }

  // ---- write normalized scores (fp32, part 2 of d_out) ----
  float* out = scores + ((size_t)bh * 2048 + s0) * 2048;
#pragma unroll
  for (int tt = 0; tt < 16; ++tt) {
    const int t = w * 256 + tt * 16 + (lane & 15);
#pragma unroll
    for (int i = 0; i < 8; ++i)
      out[(size_t)(rb + i) * 2048 + t] = acc[tt][i] * rinv[i];
  }
}

// ---------------- Kernel 3: blended = P @ V_h per (b,h) ----------------
// P fp32 [2048,2048] read back from d_out; V_h [2048,64] f16. Each block stages a 32x64
// V chunk in LDS (shared by all 8 waves), transposes it into a padded LDS tile, and reads
// B fragments as contiguous LDS vectors. Staging uses async global->LDS (double-buffered,
// s_wait_asynccnt 1 keeps chunk N+1 in flight during chunk N's WMMAs) when available.
__global__ __launch_bounds__(256) void blended_kernel(const float* __restrict__ scores,
                                                      const half_t* __restrict__ v,
                                                      half_t* __restrict__ blended) {
  const int bh = (int)blockIdx.y;
  const int b = bh >> 4, h = bh & 15;
  const int tid = (int)threadIdx.x;
  const int w = tid >> 5;
  const int s0 = (int)blockIdx.x * 128 + w * 16;
  const float* P = scores + (size_t)bh * 2048 * 2048;
  const half_t* vh = v + (size_t)b * 2048 * 1024 + h * 64;

  __shared__ half_t Vt[64][40];        // transposed chunk [n][k], padded row (80B)
#if defined(HAVE_ASYNC_TO_LDS)
  __shared__ half_t Vraw[2][32 * 64];  // double-buffered raw chunks [k][n]
#endif

  // staging role of this thread: chunk row kr (0..31), 8-half segment seg (0..7)
  const int kr = tid >> 3;
  const int seg = tid & 7;

#if defined(HAVE_ASYNC_TO_LDS)
  // global_load_async_to_lds_b128 vdst(LDS byte addr), vaddr(64b global), off
  // Low 32 bits of a generic shared pointer are the LDS offset (aperture rule).
  #define ASYNC_V_STAGE(buf, kkbase)                                                  \
    do {                                                                              \
      const half_t* _g = vh + (size_t)((kkbase) + kr) * 1024 + seg * 8;               \
      unsigned _l = (unsigned)(uintptr_t)(&Vraw[(buf)][kr * 64 + seg * 8]);           \
      asm volatile("global_load_async_to_lds_b128 %0, %1, off"                        \
                   :: "v"(_l), "v"(_g) : "memory");                                   \
    } while (0)
  ASYNC_V_STAGE(0, 0);
#endif

  v8f c[4] = {{}, {}, {}, {}};
  for (int it = 0; it < 64; ++it) {
    const int kk = it * 32;
    v8h row;
#if defined(HAVE_ASYNC_TO_LDS)
    if (it < 63) {
      ASYNC_V_STAGE((it + 1) & 1, kk + 32);
      asm volatile("s_wait_asynccnt 0x1" ::: "memory");
    } else {
      asm volatile("s_wait_asynccnt 0x0" ::: "memory");
    }
    row = *(const v8h*)&Vraw[it & 1][kr * 64 + seg * 8];
#else
    row = *(const v8h*)(vh + (size_t)(kk + kr) * 1024 + seg * 8);
#endif
    __syncthreads();                 // previous iteration's fragment reads done
#pragma unroll
    for (int j = 0; j < 8; ++j) Vt[seg * 8 + j][kr] = row[j];
    __syncthreads();                 // transposed tile ready

    v16h a = load_a_frag(P, 2048, s0, kk);
#pragma unroll
    for (int j = 0; j < 4; ++j) {
      v16h bf = load_b_lds(&Vt[0][0], 40, j * 16);
      c[j] = WMMA_F32_F16(a, bf, c[j]);
    }
  }

  const int lane = lane_id();
  const int rb = (lane & 16) ? 8 : 0;
  half_t* outp = blended + ((size_t)b * 2048 + s0) * 1024 + h * 64;
#pragma unroll
  for (int j = 0; j < 4; ++j) {
    const int col = j * 16 + (lane & 15);
#pragma unroll
    for (int i = 0; i < 8; ++i)
      outp[(size_t)(rb + i) * 1024 + col] = (half_t)c[j][i];
  }
}

// ---------------- host launch ----------------
extern "C" void kernel_launch(void* const* d_in, const int* in_sizes, int n_in,
                              void* d_out, int out_size, void* d_ws, size_t ws_size,
                              hipStream_t stream) {
  (void)in_sizes; (void)n_in; (void)out_size; (void)ws_size;
  const float* query = (const float*)d_in[0];
  const float* key_i = (const float*)d_in[1];
  const float* value = (const float*)d_in[2];
  const float* Wq = (const float*)d_in[3];
  const float* bq = (const float*)d_in[4];
  const float* Wk = (const float*)d_in[5];
  const float* bk = (const float*)d_in[6];
  const float* Wv = (const float*)d_in[7];
  const float* bv = (const float*)d_in[8];
  const float* Wo = (const float*)d_in[9];
  const float* bo = (const float*)d_in[10];

  float* out = (float*)d_out;                        // [2,2048,1024] fp32
  float* scores = out + (size_t)4096 * 1024;         // [2,16,2048,2048] fp32

  const size_t NE = (size_t)4096 * 1024;             // activation elems
  const size_t WE = (size_t)1024 * 1024;             // weight elems
  half_t* xq = (half_t*)d_ws;                        // f16 workspace (64 MB total)
  half_t* xk = xq + NE;
  half_t* xv = xk + NE;
  half_t* wq = xv + NE;
  half_t* wk = wq + WE;
  half_t* wv = wk + WE;
  half_t* wo = wv + WE;
  half_t* qh = wo + WE;
  half_t* kh = qh + NE;
  half_t* vhb = kh + NE;
  half_t* bl = vhb + NE;                             // blended

  const dim3 blk(256);
  const dim3 gcvtA((unsigned)(NE / 8 / 256));        // 2048 blocks
  const dim3 gcvtW((unsigned)(WE / 8 / 256));        // 512 blocks
  const dim3 gproj(16, 32);                          // N/64 x M/128
  const dim3 gattn(128, 32);                         // row-blocks x (B*H)
  const dim3 gblend(16, 32);                         // (S/128) x (B*H)

  cvt_f32_f16_kernel<<<gcvtA, blk, 0, stream>>>(query, xq, (int)(NE / 8));
  cvt_f32_f16_kernel<<<gcvtA, blk, 0, stream>>>(key_i, xk, (int)(NE / 8));
  cvt_f32_f16_kernel<<<gcvtA, blk, 0, stream>>>(value, xv, (int)(NE / 8));
  cvt_f32_f16_kernel<<<gcvtW, blk, 0, stream>>>(Wq, wq, (int)(WE / 8));
  cvt_f32_f16_kernel<<<gcvtW, blk, 0, stream>>>(Wk, wk, (int)(WE / 8));
  cvt_f32_f16_kernel<<<gcvtW, blk, 0, stream>>>(Wv, wv, (int)(WE / 8));
  cvt_f32_f16_kernel<<<gcvtW, blk, 0, stream>>>(Wo, wo, (int)(WE / 8));

  proj_gemm_kernel<half_t><<<gproj, blk, 0, stream>>>(xq, wq, bq, qh);
  proj_gemm_kernel<half_t><<<gproj, blk, 0, stream>>>(xk, wk, bk, kh);
  proj_gemm_kernel<half_t><<<gproj, blk, 0, stream>>>(xv, wv, bv, vhb);

  attn_softmax_kernel<<<gattn, blk, 0, stream>>>(qh, kh, scores);
  blended_kernel<<<gblend, blk, 0, stream>>>(scores, vhb, bl);
  proj_gemm_kernel<float><<<gproj, blk, 0, stream>>>(bl, wo, bo, out);
}
